// IntraModalAttention_52166672777747
// MI455X (gfx1250) — matrix operand order repaired and checked by reading, not compiled, not run on hardware
//
#include <hip/hip_runtime.h>
#include <hip/hip_bf16.h>

// ---------------------------------------------------------------------------
// IntraModalAttention on MI455X (gfx1250), wave32 + WMMA bf16.
// Sizes: B=4 T=32 D=768 NH=12 HD=64 Hp=Wp=16 S=256, BS=B*S=1024, M=BS*T=32768
// ---------------------------------------------------------------------------

typedef __bf16 v8bf  __attribute__((ext_vector_type(8)));
typedef __bf16 v16bf __attribute__((ext_vector_type(16)));
typedef float  v8f   __attribute__((ext_vector_type(8)));

__device__ __forceinline__ v16bf cat8(v8bf lo, v8bf hi) {
  return __builtin_shufflevector(lo, hi, 0,1,2,3,4,5,6,7,8,9,10,11,12,13,14,15);
}
__device__ __forceinline__ v8f vzero8() {
  v8f z;
#pragma unroll
  for (int i = 0; i < 8; ++i) z[i] = 0.0f;
  return z;
}
__device__ __forceinline__ v8f wmma_bf16(v16bf a, v16bf b, v8f c) {
  return __builtin_amdgcn_wmma_f32_16x16x32_bf16(false, a, false, b, (short)0, c,
                                                 false, false);
}

// ---------------------------------------------------------------------------
// Kernel 1: weights fp32 -> bf16. wqkv = [Wq;Wk;Wv] (2304x768), wob = Wo.
// ---------------------------------------------------------------------------
__global__ __launch_bounds__(256) void pack_w_kernel(
    const float* __restrict__ wq, const float* __restrict__ wk,
    const float* __restrict__ wv, const float* __restrict__ wo,
    __bf16* __restrict__ wqkv, __bf16* __restrict__ wob) {
  int i = blockIdx.x * 256 + threadIdx.x;       // 768*768 = 589824 elems
  if (i < 589824) {
    wqkv[i]            = (__bf16)wq[i];
    wqkv[589824 + i]   = (__bf16)wk[i];
    wqkv[1179648 + i]  = (__bf16)wv[i];
    wob[i]             = (__bf16)wo[i];
  }
}

// ---------------------------------------------------------------------------
// Kernel 2: h (B,T,D,Hp,Wp) fp32 -> hseq (M=32768, K=768) bf16 row-major.
// hseq[((b*256+p)*32+t)*768 + d] = h[((b*32+t)*768+d)*256 + p]
// LDS 32x(256+1) fp32 tile: coalesced reads over p, 64B-chunk writes over d.
// grid = (24 d-chunks, 128 bt), block = 256 (= p)
// ---------------------------------------------------------------------------
__global__ __launch_bounds__(256) void pack_h_kernel(
    const float* __restrict__ h, __bf16* __restrict__ hseq) {
  __shared__ float tile[32][257];
  const int bt = blockIdx.y;
  const int d0 = blockIdx.x * 32;
  const int p  = threadIdx.x;
#pragma unroll 4
  for (int dd = 0; dd < 32; ++dd)
    tile[dd][p] = h[((size_t)bt * 768 + d0 + dd) * 256 + p];
  __syncthreads();
  const int b = bt >> 5, t = bt & 31;
  __bf16* dst = hseq + ((size_t)(b * 256 + p) * 32 + t) * 768 + d0;
#pragma unroll 4
  for (int dd = 0; dd < 32; ++dd) dst[dd] = (__bf16)tile[dd][p];
}

// ---------------------------------------------------------------------------
// Kernel 3: u_pen[(b*256+p)*32+t] = lam[bt*256+p] * mean_d u[(bt*768+d)*256+p]
// grid = 128 (bt), block = 256 (p); fully coalesced reduction loads.
// ---------------------------------------------------------------------------
__global__ __launch_bounds__(256) void upen_kernel(
    const float* __restrict__ u, const float* __restrict__ lam,
    float* __restrict__ upen) {
  const int bt = blockIdx.x;
  const int p  = threadIdx.x;
  const float* src = u + (size_t)bt * 768 * 256 + p;
  float acc = 0.0f;
  for (int d = 0; d < 768; ++d) acc += src[(size_t)d * 256];
  const int b = bt >> 5, t = bt & 31;
  upen[(b * 256 + p) * 32 + t] = lam[(size_t)bt * 256 + p] * (acc * (1.0f / 768.0f));
}

// ---------------------------------------------------------------------------
// Kernel 4: QKV GEMM.  C(32768x2304) = hseq(32768x768) x W^T, bias added.
// Wave tile 64x64 (16 accum fragments). 8 waves/WG -> WG tile 128x256.
// Epilogue scatters: Q,K as (seq,head,t,hd), V transposed as (seq,head,hd,t).
// region/head are computed from the SCALAR tile base (16-aligned boundaries),
// so the Q/K/V selection is a uniform branch, no per-lane division, EXEC
// stays fully converged.
// grid = (2304/256=9, 32768/128=256), block = 256.
// ---------------------------------------------------------------------------
__global__ __launch_bounds__(256) void gemm_qkv_kernel(
    const __bf16* __restrict__ A, const __bf16* __restrict__ Bw,
    const float* __restrict__ bq, const float* __restrict__ bk,
    const float* __restrict__ bv,
    __bf16* __restrict__ qout, __bf16* __restrict__ kout,
    __bf16* __restrict__ vtout) {
  const int lane = threadIdx.x & 31;
  const int wave = threadIdx.x >> 5;
  const int wm = wave >> 2, wn = wave & 3;
  const int m0 = blockIdx.y * 128 + wm * 64;
  const int j0 = blockIdx.x * 256 + wn * 64;
  const int lm = lane & 15, lh = lane >> 4;

  v8f acc[4][4];
#pragma unroll
  for (int mt = 0; mt < 4; ++mt)
#pragma unroll
    for (int nt = 0; nt < 4; ++nt) acc[mt][nt] = vzero8();

  for (int kb = 0; kb < 24; ++kb) {        // K = 768 = 24 * 32
    v16bf fa[4], fb[4];
#pragma unroll
    for (int mt = 0; mt < 4; ++mt) {       // A frag: lane holds row m, K-contig
      const __bf16* pa = A + (size_t)(m0 + mt * 16 + lm) * 768 + kb * 32 + lh * 8;
      fa[mt] = cat8(*(const v8bf*)pa, *(const v8bf*)(pa + 16));
    }
#pragma unroll
    for (int nt = 0; nt < 4; ++nt) {       // B frag: lane holds col j = W row j
      const __bf16* pb = Bw + (size_t)(j0 + nt * 16 + lm) * 768 + kb * 32 + lh * 8;
      fb[nt] = cat8(*(const v8bf*)pb, *(const v8bf*)(pb + 16));
    }
#pragma unroll
    for (int mt = 0; mt < 4; ++mt)
#pragma unroll
      for (int nt = 0; nt < 4; ++nt)
        acc[mt][nt] = wmma_bf16(fa[mt], fb[nt], acc[mt][nt]);
  }

#pragma unroll
  for (int nt = 0; nt < 4; ++nt) {
    const int jt = j0 + nt * 16;           // uniform tile base column
    const int region = jt / 768;           // scalar: 0=Q 1=K 2=V
    const int jjb = jt - region * 768;     // scalar, 16-aligned
    const int head = jjb >> 6;             // scalar (64 % 16 == 0)
    const int hd = (jjb & 63) + lm;        // per-lane head-dim index
    const float bias = (region == 0) ? bq[jjb + lm]
                     : (region == 1) ? bk[jjb + lm] : bv[jjb + lm];
    if (region < 2) {
      __bf16* __restrict__ dst = (region == 0) ? qout : kout;
#pragma unroll
      for (int mt = 0; mt < 4; ++mt)
#pragma unroll
        for (int r = 0; r < 8; ++r) {
          const int m = m0 + mt * 16 + lh * 8 + r;   // C layout row
          const int nseq = m >> 5, t = m & 31;
          dst[(size_t)((nseq * 12 + head) * 32 + t) * 64 + hd] =
              (__bf16)(acc[mt][nt][r] + bias);
        }
    } else {
#pragma unroll
      for (int mt = 0; mt < 4; ++mt)
#pragma unroll
        for (int r = 0; r < 8; ++r) {
          const int m = m0 + mt * 16 + lh * 8 + r;
          const int nseq = m >> 5, t = m & 31;
          vtout[(size_t)((nseq * 12 + head) * 64 + hd) * 32 + t] =
              (__bf16)(acc[mt][nt][r] + bias);
        }
    }
  }
}

// ---------------------------------------------------------------------------
// Kernel 5: attention. One wave per (seq, head) unit; 4 waves per block.
// scores = Q K^T / 8 - u_pen[key];  softmax over 32 keys (shfl_xor within
// 16-lane halves matches the WMMA C layout);  out = attn x V via WMMA after
// an LDS transpose of the probabilities into A-fragment order.
// grid = 12288/4 = 3072, block = 128.
// ---------------------------------------------------------------------------
__global__ __launch_bounds__(128) void attn_kernel(
    const __bf16* __restrict__ Q, const __bf16* __restrict__ K,
    const __bf16* __restrict__ Vt, const float* __restrict__ upen,
    __bf16* __restrict__ aout /* (32768, 768) bf16 row-major */) {
  __shared__ float smem[4][32 * 32];
  const int lane = threadIdx.x & 31;
  const int wave = threadIdx.x >> 5;
  const int unit = blockIdx.x * 4 + wave;  // 0..12287
  const int n = unit / 12, head = unit % 12;
  const int lm = lane & 15, lh = lane >> 4;

  const __bf16* qb = Q  + (size_t)((n * 12 + head) * 32) * 64;
  const __bf16* kp = K  + (size_t)((n * 12 + head) * 32) * 64;
  const __bf16* vb = Vt + (size_t)((n * 12 + head) * 64) * 32;

  v16bf fq[2][2], fk[2][2];
#pragma unroll
  for (int mt = 0; mt < 2; ++mt)
#pragma unroll
    for (int kt = 0; kt < 2; ++kt) {
      const __bf16* p = qb + (size_t)(mt * 16 + lm) * 64 + kt * 32 + lh * 8;
      fq[mt][kt] = cat8(*(const v8bf*)p, *(const v8bf*)(p + 16));
    }
#pragma unroll
  for (int nt = 0; nt < 2; ++nt)
#pragma unroll
    for (int kt = 0; kt < 2; ++kt) {
      const __bf16* p = kp + (size_t)(nt * 16 + lm) * 64 + kt * 32 + lh * 8;
      fk[nt][kt] = cat8(*(const v8bf*)p, *(const v8bf*)(p + 16));
    }

  v8f sc[2][2];
#pragma unroll
  for (int mt = 0; mt < 2; ++mt)
#pragma unroll
    for (int nt = 0; nt < 2; ++nt) {
      v8f c = vzero8();
      c = wmma_bf16(fq[mt][0], fk[nt][0], c);
      c = wmma_bf16(fq[mt][1], fk[nt][1], c);
      sc[mt][nt] = c;
    }

  const float pen0 = upen[n * 32 + lm];
  const float pen1 = upen[n * 32 + 16 + lm];

#pragma unroll
  for (int mt = 0; mt < 2; ++mt) {
#pragma unroll
    for (int r = 0; r < 8; ++r) {
      float s0 = sc[mt][0][r] * 0.125f - pen0;   // col j = lm
      float s1 = sc[mt][1][r] * 0.125f - pen1;   // col j = 16+lm
      float mx = fmaxf(s0, s1);
#pragma unroll
      for (int o = 1; o < 16; o <<= 1) mx = fmaxf(mx, __shfl_xor(mx, o, 32));
      const float e0 = __expf(s0 - mx);
      const float e1 = __expf(s1 - mx);
      float sum = e0 + e1;
#pragma unroll
      for (int o = 1; o < 16; o <<= 1) sum += __shfl_xor(sum, o, 32);
      const float inv = 1.0f / sum;
      const int m = mt * 16 + lh * 8 + r;        // query row
      smem[wave][m * 32 + lm]      = e0 * inv;
      smem[wave][m * 32 + 16 + lm] = e1 * inv;
    }
  }
  __syncthreads();

  v16bf fa[2];
#pragma unroll
  for (int mt = 0; mt < 2; ++mt) {               // A frag of probs from LDS
    const float* p = &smem[wave][(mt * 16 + lm) * 32 + lh * 8];
    v16bf a;
#pragma unroll
    for (int e = 0; e < 8; ++e) {
      a[e]     = (__bf16)p[e];
      a[e + 8] = (__bf16)p[16 + e];
    }
    fa[mt] = a;
  }
  v16bf fv[4];
#pragma unroll
  for (int nt = 0; nt < 4; ++nt) {               // B frag of V from Vt (key-contig)
    const __bf16* p = vb + (size_t)(nt * 16 + lm) * 32 + lh * 8;
    fv[nt] = cat8(*(const v8bf*)p, *(const v8bf*)(p + 16));
  }
#pragma unroll
  for (int mt = 0; mt < 2; ++mt)
#pragma unroll
    for (int nt = 0; nt < 4; ++nt) {
      v8f o = vzero8();
      o = wmma_bf16(fa[mt], fv[nt], o);
      const int hd = nt * 16 + lm;
#pragma unroll
      for (int r = 0; r < 8; ++r) {
        const int t = mt * 16 + lh * 8 + r;
        aout[(size_t)(n * 32 + t) * 768 + head * 64 + hd] = (__bf16)o[r];
      }
    }
}

// ---------------------------------------------------------------------------
// Kernel 6: output GEMM. Ctmp(32768x768 f32) = aout x Wo^T + bo.
// grid = (768/256=3, 256), block = 256.
// ---------------------------------------------------------------------------
__global__ __launch_bounds__(256) void gemm_o_kernel(
    const __bf16* __restrict__ A, const __bf16* __restrict__ Bw,
    const float* __restrict__ bo, float* __restrict__ Ctmp) {
  const int lane = threadIdx.x & 31;
  const int wave = threadIdx.x >> 5;
  const int wm = wave >> 2, wn = wave & 3;
  const int m0 = blockIdx.y * 128 + wm * 64;
  const int j0 = blockIdx.x * 256 + wn * 64;
  const int lm = lane & 15, lh = lane >> 4;

  v8f acc[4][4];
#pragma unroll
  for (int mt = 0; mt < 4; ++mt)
#pragma unroll
    for (int nt = 0; nt < 4; ++nt) acc[mt][nt] = vzero8();

  for (int kb = 0; kb < 24; ++kb) {
    v16bf fa[4], fb[4];
#pragma unroll
    for (int mt = 0; mt < 4; ++mt) {
      const __bf16* pa = A + (size_t)(m0 + mt * 16 + lm) * 768 + kb * 32 + lh * 8;
      fa[mt] = cat8(*(const v8bf*)pa, *(const v8bf*)(pa + 16));
    }
#pragma unroll
    for (int nt = 0; nt < 4; ++nt) {
      const __bf16* pb = Bw + (size_t)(j0 + nt * 16 + lm) * 768 + kb * 32 + lh * 8;
      fb[nt] = cat8(*(const v8bf*)pb, *(const v8bf*)(pb + 16));
    }
#pragma unroll
    for (int mt = 0; mt < 4; ++mt)
#pragma unroll
      for (int nt = 0; nt < 4; ++nt)
        acc[mt][nt] = wmma_bf16(fa[mt], fb[nt], acc[mt][nt]);
  }

#pragma unroll
  for (int mt = 0; mt < 4; ++mt)
#pragma unroll
    for (int nt = 0; nt < 4; ++nt) {
      const int j = j0 + nt * 16 + lm;
      const float bias = bo[j];
#pragma unroll
      for (int r = 0; r < 8; ++r) {
        const int m = m0 + mt * 16 + lh * 8 + r;
        Ctmp[(size_t)m * 768 + j] = acc[mt][nt][r] + bias;
      }
    }
}

// ---------------------------------------------------------------------------
// Kernel 7: Ctmp (M=32768 x 768, seq-major) -> out (B,T,D,Hp,Wp) fp32.
// LDS transpose so final HBM stores are fully coalesced over p.
// grid = (24, 128), block = 256 (= p).
// ---------------------------------------------------------------------------
__global__ __launch_bounds__(256) void unpack_out_kernel(
    const float* __restrict__ Ctmp, float* __restrict__ out) {
  __shared__ float tile[32][257];
  const int bt = blockIdx.y;
  const int d0 = blockIdx.x * 32;
  const int p  = threadIdx.x;
  const int b = bt >> 5, t = bt & 31;
  const float* src = Ctmp + ((size_t)(b * 256 + p) * 32 + t) * 768 + d0;
#pragma unroll 4
  for (int dd = 0; dd < 32; ++dd) tile[dd][p] = src[dd];
  __syncthreads();
#pragma unroll 4
  for (int dd = 0; dd < 32; ++dd)
    out[((size_t)bt * 768 + d0 + dd) * 256 + p] = tile[dd][p];
}

// ---------------------------------------------------------------------------
// Host launcher. Workspace layout (bytes), with dead-buffer aliasing:
//   [0]          hseq   bf16  50,331,648   (dead after GEMM1 -> reused as aout)
//   [50331648]   wqkv   bf16   3,538,944
//   [53870592]   wob    bf16   1,179,648
//   [55050240]   q      bf16  50,331,648   \  dead after attn -> Ctmp f32
//   [105381888]  k      bf16  50,331,648   /  (100,663,296 B exactly)
//   [155713536]  vt     bf16  50,331,648
//   [206045184]  upen   f32      131,072     -> total 206,176,256 B
// ---------------------------------------------------------------------------
extern "C" void kernel_launch(void* const* d_in, const int* in_sizes, int n_in,
                              void* d_out, int out_size, void* d_ws, size_t ws_size,
                              hipStream_t stream) {
  const float* h    = (const float*)d_in[0];
  const float* u    = (const float*)d_in[1];
  const float* lam  = (const float*)d_in[2];
  const float* Wq   = (const float*)d_in[3];
  const float* bq   = (const float*)d_in[4];
  const float* Wk   = (const float*)d_in[5];
  const float* bk   = (const float*)d_in[6];
  const float* Wv   = (const float*)d_in[7];
  const float* bv   = (const float*)d_in[8];
  const float* Wo   = (const float*)d_in[9];
  const float* bo   = (const float*)d_in[10];
  float* out = (float*)d_out;

  char* ws = (char*)d_ws;
  __bf16* hseq = (__bf16*)(ws);
  __bf16* wqkv = (__bf16*)(ws + 50331648);
  __bf16* wob  = (__bf16*)(ws + 53870592);
  __bf16* qb   = (__bf16*)(ws + 55050240);
  __bf16* kb   = (__bf16*)(ws + 105381888);
  __bf16* vtb  = (__bf16*)(ws + 155713536);
  float*  upen = (float*)(ws + 206045184);
  __bf16* aout = hseq;                 // reuse: hseq dead after GEMM1
  float*  Ctmp = (float*)qb;           // reuse: q,k dead after attention

  pack_w_kernel<<<2304, 256, 0, stream>>>(Wq, Wk, Wv, Wo, wqkv, wob);
  pack_h_kernel<<<dim3(24, 128), 256, 0, stream>>>(h, hseq);
  upen_kernel<<<128, 256, 0, stream>>>(u, lam, upen);
  gemm_qkv_kernel<<<dim3(9, 256), 256, 0, stream>>>(hseq, wqkv, bq, bk, bv,
                                                    qb, kb, vtb);
  attn_kernel<<<3072, 128, 0, stream>>>(qb, kb, vtb, upen, aout);
  gemm_o_kernel<<<dim3(3, 256), 256, 0, stream>>>(aout, wob, bo, Ctmp);
  unpack_out_kernel<<<dim3(24, 128), 256, 0, stream>>>(Ctmp, out);
}
